// SWAN_14465449853267
// MI455X (gfx1250) — matrix operand order repaired and checked by reading, not compiled, not run on hardware
//
#include <hip/hip_runtime.h>
#include <hip/hip_bf16.h>

// ---------------------------------------------------------------------------
// Problem constants (from the reference)
// ---------------------------------------------------------------------------
#define B_    4
#define T1_   16
#define T2_   16
#define SEG_  4
#define V_    8000
#define E_    256
#define H_    256
#define ENC_  512
#define EOS_  2
#define NEG_  (-1e30f)
#define NST   1088      // B*T1*(T2+1)

typedef __attribute__((ext_vector_type(16))) _Float16 v16h;
typedef __attribute__((ext_vector_type(8)))  _Float16 v8h;
typedef __attribute__((ext_vector_type(8)))  float    v8f;

// ---------------------------------------------------------------------------
// WMMA helper: D = A(16x32 f16) * B(32x16 f16) + C (f32)
// ---------------------------------------------------------------------------
__device__ __forceinline__ v8f wmma_f16(v16h a, v16h b, v8f c) {
  return __builtin_amdgcn_wmma_f32_16x16x32_f16(
      /*neg_a=*/false, a, /*neg_b=*/false, b,
      /*c_mod=*/(short)0, c, /*reuse_a=*/false, /*reuse_b=*/false);
}

// A-fragment (16x32, f16) from row-major f16 LDS buffer.
// This code's k order (matches pack_kcat/pack_ncat): lane L holds row M = L%16; with g = L/16,
// halves 0..15 = K in [koff + g*16, koff + g*16 + 16)
__device__ __forceinline__ v16h load_afrag(const _Float16* base, int strideH,
                                           int koff, int lane) {
  int g = lane >> 4, m = lane & 15;
  const _Float16* p = base + m * strideH + koff + g * 16;
  v8h lo = *(const v8h*)(p);
  v8h hi = *(const v8h*)(p + 8);
  v16h a;
#pragma unroll
  for (int t = 0; t < 8; ++t) { a[t] = lo[t]; a[t + 8] = hi[t]; }
  return a;
}

// B-fragment from pre-packed weights: one contiguous 32B load per lane.
__device__ __forceinline__ v16h load_bfrag(const _Float16* w, int frag, int lane) {
  return *(const v16h*)(w + (size_t)frag * 512 + lane * 16);
}

__device__ __forceinline__ float sigm(float x) { return 1.f / (1.f + __expf(-x)); }

// ---------------------------------------------------------------------------
// Weight packing into WMMA B-fragment layout (contiguous k order, same as load_afrag):
// fragment (kt,nt); slot(lane,h): K = kt*32 + (lane/16)*16 + h, N = nt*16 + lane%16
// Virtual W is [N][K] row-major; B(k,n) = W[n][k]  (since out = act @ W^T).
// ---------------------------------------------------------------------------
__global__ void pack_kcat(_Float16* dst, const float* A, const float* Bm,
                          int Ka, int Kb, int N) {
  int K = Ka + Kb;
  int ntiles = N >> 4;
  int total = (K >> 5) * ntiles * 32;
  int idx = blockIdx.x * blockDim.x + threadIdx.x;
  if (idx >= total) return;
  int lane = idx & 31, frag = idx >> 5;
  int kt = frag / ntiles, nt = frag % ntiles;
  int n = nt * 16 + (lane & 15);
  int kbase = kt * 32 + ((lane >> 4) << 4);
  v16h o;
#pragma unroll
  for (int h = 0; h < 16; ++h) {
    int k = kbase + h;
    float v = (k < Ka) ? A[(size_t)n * Ka + k] : Bm[(size_t)n * Kb + (k - Ka)];
    o[h] = (_Float16)v;
  }
  *(v16h*)(dst + (size_t)frag * 512 + lane * 16) = o;
}

__global__ void pack_ncat(_Float16* dst, const float* A, const float* Bm,
                          int K, int Nhalf) {
  int ntiles = (2 * Nhalf) >> 4;
  int total = (K >> 5) * ntiles * 32;
  int idx = blockIdx.x * blockDim.x + threadIdx.x;
  if (idx >= total) return;
  int lane = idx & 31, frag = idx >> 5;
  int kt = frag / ntiles, nt = frag % ntiles;
  int n = nt * 16 + (lane & 15);
  int kbase = kt * 32 + ((lane >> 4) << 4);
  v16h o;
#pragma unroll
  for (int h = 0; h < 16; ++h) {
    int k = kbase + h;
    float v = (n < Nhalf) ? A[(size_t)n * K + k] : Bm[(size_t)(n - Nhalf) * K + k];
    o[h] = (_Float16)v;
  }
  *(v16h*)(dst + (size_t)frag * 512 + lane * 16) = o;
}

// ---------------------------------------------------------------------------
// Init: h0 = eo @ proj_h_w.T + b_h ; c0 = eo @ proj_c_w.T + b_c
// 64 rows (cb = b*16 + t1), K=ENC=512, N=512 (h | c). 4 blocks x 256 threads.
// ---------------------------------------------------------------------------
__global__ void __launch_bounds__(256, 1)
init_proj_kernel(const float* __restrict__ enc,
                 const float* __restrict__ phb,
                 const float* __restrict__ pcb,
                 const _Float16* __restrict__ Wpj,
                 _Float16* __restrict__ ctx_ws,
                 float* __restrict__ c0_ws) {
  __shared__ __attribute__((aligned(32))) _Float16 sA[16 * 520];
  int tid = threadIdx.x, lane = tid & 31, w = tid >> 5;
  int blk = blockIdx.x;  // == batch b ; row m == t1
  // stage eo rows to LDS as f16: enc is (T1,B,ENC)
#pragma unroll 4
  for (int it = 0; it < 32; ++it) {
    int e = tid * 32 + it;
    int m = e >> 9, k = e & 511;
    sA[m * 520 + k] = (_Float16)enc[((size_t)m * B_ + blk) * ENC_ + k];
  }
  __syncthreads();
#pragma unroll
  for (int q = 0; q < 4; ++q) {
    int nt = w * 4 + q;            // 0..31
    v8f acc = {};
    for (int kt = 0; kt < 16; ++kt) {
      v16h a = load_afrag(sA, 520, kt * 32, lane);
      v16h b = load_bfrag(Wpj, kt * 32 + nt, lane);
      acc = wmma_f16(a, b, acc);
    }
    int cl = (nt & 15) * 16 + (lane & 15);   // column within H=256 block
    bool isH = nt < 16;
    float bias = isH ? phb[cl] : pcb[cl];
    int grb = blk * 16;
#pragma unroll
    for (int r = 0; r < 8; ++r) {
      int m = ((lane >> 4) << 3) + r;
      float v = acc[r] + bias;
      if (isH) ctx_ws[(size_t)(grb + m) * H_ + cl] = (_Float16)v;
      else     c0_ws[(size_t)(grb + m) * H_ + cl] = v;
    }
  }
}

// ---------------------------------------------------------------------------
// Main decode kernel: 68 blocks x 16 states, 256 threads (8 wave32).
// Runs all 5 scan steps; emits sel/eos log-probs per (step, state).
// ---------------------------------------------------------------------------
#define ST0 776   // [ctx(256) | x(256) | h1(256)] + 8 pad halves
#define ST2 264   // h2(256) + 8 pad halves

__global__ void __launch_bounds__(256, 1)
decode_kernel(const int* __restrict__ ytok,
              const float* __restrict__ embed,
              const float* __restrict__ start_embed,
              const float* __restrict__ b0,
              const float* __restrict__ b1,
              const float* __restrict__ fcb,
              const _Float16* __restrict__ Wp0,
              const _Float16* __restrict__ Wp1,
              const _Float16* __restrict__ Wfc,
              const _Float16* __restrict__ ctx_ws,
              const float* __restrict__ c0_ws,
              float* __restrict__ sel_ws,
              float* __restrict__ eos_ws) {
  __shared__ __attribute__((aligned(32))) _Float16 sAct[16 * ST0];
  __shared__ __attribute__((aligned(32))) _Float16 sH2[16 * ST2];
  __shared__ float sRm[8][16], sRs[8][16], sSel[8][16], sEos[8][16];
  __shared__ int sTgt[16], sB[16], sI[16], sCb[16];

  int tid = threadIdx.x, lane = tid & 31, w = tid >> 5;
  int g = lane >> 4, ln = lane & 15;

  if (tid < 16) {
    int n = blockIdx.x * 16 + tid;
    int b = n / (T1_ * 17), rem = n % (T1_ * 17);
    sB[tid] = b; sI[tid] = rem % 17; sCb[tid] = b * T1_ + rem / 17;
  }
  __syncthreads();

  // ctx -> sAct[:,0:256], h1 init = ctx -> sAct[:,512:768], h2 = 0
  {
    int m = tid >> 4, e0 = (tid & 15) * 16;
#pragma unroll
    for (int e = 0; e < 16; ++e) {
      int col = e0 + e;
      _Float16 cv = ctx_ws[(size_t)sCb[m] * H_ + col];
      sAct[m * ST0 + col] = cv;
      sAct[m * ST0 + 512 + col] = cv;
      sH2[m * ST2 + col] = (_Float16)0.f;
    }
  }
  __syncthreads();

  // Cell state lives in registers in the WMMA C/D layout:
  // wave w owns hidden-column tiles jt = 2w, 2w+1.
  v8f c1r[2], c2r[2];
#pragma unroll
  for (int q = 0; q < 2; ++q) {
    int jt = w * 2 + q, j = jt * 16 + ln;
#pragma unroll
    for (int r = 0; r < 8; ++r) {
      int m = (g << 3) + r;
      c1r[q][r] = c0_ws[(size_t)sCb[m] * H_ + j];
      c2r[q][r] = 0.f;
    }
  }

  for (int s = 0; s < SEG_ + 1; ++s) {
    // ---- build x (embedding) and targets ----
    {
      int m = tid >> 4, e0 = (tid & 15) * 16;
      int b = sB[m], i = sI[m];
      const float* src;
      if (s == 0) src = start_embed;
      else {
        int tk = ytok[b * T2_ + min(i + s - 1, T2_ - 1)];
        src = embed + (size_t)tk * E_;
      }
#pragma unroll
      for (int e = 0; e < 16; ++e)
        sAct[m * ST0 + 256 + e0 + e] = (_Float16)src[e0 + e];
      if (tid < 16) {
        int bb = sB[tid], ii = sI[tid];
        sTgt[tid] = (s < SEG_) ? ytok[bb * T2_ + min(ii + s, T2_ - 1)] : 0;
      }
    }
    __syncthreads();

    // ---- LSTM cell 0: A = [ctx|x|h1] (K=768), W0cat (N=1024) ----
    // Load each A fragment once; issue 8 WMMAs (2 j-tiles x 4 gates) per load.
    v8f acc[2][4];
    {
      v8f z = {};
#pragma unroll
      for (int q = 0; q < 2; ++q)
#pragma unroll
        for (int t = 0; t < 4; ++t) acc[q][t] = z;
    }
    for (int kt = 0; kt < 24; ++kt) {
      v16h a = load_afrag(sAct, ST0, kt * 32, lane);
#pragma unroll
      for (int q = 0; q < 2; ++q) {
        int jt = w * 2 + q;
#pragma unroll
        for (int gate = 0; gate < 4; ++gate)
          acc[q][gate] = wmma_f16(a, load_bfrag(Wp0, kt * 64 + gate * 16 + jt, lane),
                                  acc[q][gate]);
      }
    }
    __syncthreads();   // all waves done reading h1 before overwrite
#pragma unroll
    for (int q = 0; q < 2; ++q) {
      int jt = w * 2 + q, j = jt * 16 + ln;
      float bi = b0[j], bf = b0[256 + j], bg = b0[512 + j], bo = b0[768 + j];
#pragma unroll
      for (int r = 0; r < 8; ++r) {
        float gi = acc[q][0][r] + bi, gf = acc[q][1][r] + bf;
        float gg = acc[q][2][r] + bg, go = acc[q][3][r] + bo;
        float cn = sigm(gf) * c1r[q][r] + sigm(gi) * tanhf(gg);
        float hn = sigm(go) * tanhf(cn);
        c1r[q][r] = cn;
        int m = (g << 3) + r;
        sAct[m * ST0 + 512 + j] = (_Float16)hn;
      }
    }
    __syncthreads();

    // ---- LSTM cell 1: A = [h1|h2] (K=512), W1cat (N=1024) ----
    {
      v8f z = {};
#pragma unroll
      for (int q = 0; q < 2; ++q)
#pragma unroll
        for (int t = 0; t < 4; ++t) acc[q][t] = z;
    }
    for (int kt = 0; kt < 16; ++kt) {
      v16h a = (kt < 8) ? load_afrag(sAct, ST0, 512 + kt * 32, lane)
                        : load_afrag(sH2, ST2, (kt - 8) * 32, lane);
#pragma unroll
      for (int q = 0; q < 2; ++q) {
        int jt = w * 2 + q;
#pragma unroll
        for (int gate = 0; gate < 4; ++gate)
          acc[q][gate] = wmma_f16(a, load_bfrag(Wp1, kt * 64 + gate * 16 + jt, lane),
                                  acc[q][gate]);
      }
    }
    __syncthreads();   // all waves done reading h2 before overwrite
#pragma unroll
    for (int q = 0; q < 2; ++q) {
      int jt = w * 2 + q, j = jt * 16 + ln;
      float bi = b1[j], bf = b1[256 + j], bg = b1[512 + j], bo = b1[768 + j];
#pragma unroll
      for (int r = 0; r < 8; ++r) {
        float gi = acc[q][0][r] + bi, gf = acc[q][1][r] + bf;
        float gg = acc[q][2][r] + bg, go = acc[q][3][r] + bo;
        float cn = sigm(gf) * c2r[q][r] + sigm(gi) * tanhf(gg);
        float hn = sigm(go) * tanhf(cn);
        c2r[q][r] = cn;
        int m = (g << 3) + r;
        sH2[m * ST2 + j] = (_Float16)hn;
      }
    }
    __syncthreads();

    // ---- fc + fused log-softmax: A = h2 (K=256), W = fc_w (N=8000) ----
    // Hoist the 8 A fragments into registers: h2 is invariant over the V sweep.
    v16h afr[8];
#pragma unroll
    for (int kt = 0; kt < 8; ++kt) afr[kt] = load_afrag(sH2, ST2, kt * 32, lane);

    float mr[8], sr[8], se[8], eo[8];
    int tg[8];
#pragma unroll
    for (int r = 0; r < 8; ++r) {
      mr[r] = NEG_; sr[r] = 0.f; se[r] = NEG_; eo[r] = NEG_;
      tg[r] = sTgt[(g << 3) + r];
    }
    for (int ct = w; ct < V_ / 16; ct += 8) {
      v8f a8 = {};
#pragma unroll
      for (int kt = 0; kt < 8; ++kt)
        a8 = wmma_f16(afr[kt], load_bfrag(Wfc, kt * (V_ / 16) + ct, lane), a8);
      int j = ct * 16 + ln;
      float bias = fcb[j];
#pragma unroll
      for (int r = 0; r < 8; ++r) {
        float v = a8[r] + bias;
        if (v > mr[r]) { sr[r] = sr[r] * __expf(mr[r] - v) + 1.f; mr[r] = v; }
        else           { sr[r] += __expf(v - mr[r]); }
        if (j == tg[r]) se[r] = v;
        if (j == EOS_)  eo[r] = v;
      }
    }
    // reduce across the 16 lanes of each half-wave
#pragma unroll
    for (int off = 1; off < 16; off <<= 1) {
#pragma unroll
      for (int r = 0; r < 8; ++r) {
        float mo = __shfl_xor(mr[r], off, 32);
        float so = __shfl_xor(sr[r], off, 32);
        float MM = fmaxf(mr[r], mo);
        sr[r] = sr[r] * __expf(mr[r] - MM) + so * __expf(mo - MM);
        mr[r] = MM;
        se[r] = fmaxf(se[r], __shfl_xor(se[r], off, 32));
        eo[r] = fmaxf(eo[r], __shfl_xor(eo[r], off, 32));
      }
    }
    if (ln == 0) {
#pragma unroll
      for (int r = 0; r < 8; ++r) {
        int m = (g << 3) + r;
        sRm[w][m] = mr[r]; sRs[w][m] = sr[r];
        sSel[w][m] = se[r]; sEos[w][m] = eo[r];
      }
    }
    __syncthreads();
    if (tid < 16) {
      float M = NEG_, S = 0.f, SE = NEG_, EO = NEG_;
#pragma unroll
      for (int w2 = 0; w2 < 8; ++w2) {
        float m2 = sRm[w2][tid], s2 = sRs[w2][tid];
        float MM = fmaxf(M, m2);
        S = S * __expf(M - MM) + s2 * __expf(m2 - MM);
        M = MM;
        SE = fmaxf(SE, sSel[w2][tid]);
        EO = fmaxf(EO, sEos[w2][tid]);
      }
      float lz = M + __logf(S);
      int n = blockIdx.x * 16 + tid;
      sel_ws[s * NST + n] = SE - lz;
      eos_ws[s * NST + n] = EO - lz;
    }
    __syncthreads();
  }
}

// ---------------------------------------------------------------------------
// Final DP over (T1, T2+1, SEG+1): alpha_new[ii] = LSE_j alpha[ii-j] + logpy(ii-j, j)
// ---------------------------------------------------------------------------
__global__ void dp_kernel(const float* __restrict__ sel_ws,
                          const float* __restrict__ eos_ws,
                          float* __restrict__ out) {
  __shared__ float alpha[B_][T2_ + 1];
  int t = threadIdx.x;
  int b = t / 17, ii = t % 17;
  bool act = t < B_ * 17;
  if (act) alpha[b][ii] = (ii == 0) ? 0.f : NEG_;
  __syncthreads();
  for (int t1 = 0; t1 < T1_; ++t1) {
    float vnew = NEG_;
    if (act) {
      float vals[SEG_ + 1];
      float M = NEG_;
#pragma unroll
      for (int j = 0; j <= SEG_; ++j) {
        float v = NEG_;
        int i = ii - j;
        if (i >= 0) {
          int n = (b * T1_ + t1) * 17 + i;
          float cum = 0.f;
          for (int ss = 0; ss < j; ++ss) cum += sel_ws[ss * NST + n];
          v = alpha[b][i] + cum + eos_ws[j * NST + n];
        }
        vals[j] = v;
        M = fmaxf(M, v);
      }
      if (M > -1e29f) {
        float S = 0.f;
#pragma unroll
        for (int j = 0; j <= SEG_; ++j) S += __expf(vals[j] - M);
        vnew = M + __logf(S);
      }
    }
    __syncthreads();
    if (act) alpha[b][ii] = vnew;
    __syncthreads();
  }
  if (act && ii == T2_) out[b] = alpha[b][T2_];
}

// ---------------------------------------------------------------------------
// Launch
// ---------------------------------------------------------------------------
extern "C" void kernel_launch(void* const* d_in, const int* in_sizes, int n_in,
                              void* d_out, int out_size, void* d_ws, size_t ws_size,
                              hipStream_t stream) {
  (void)in_sizes; (void)n_in; (void)out_size; (void)ws_size;
  const int*   ytok      = (const int*)d_in[0];
  const float* enc       = (const float*)d_in[1];
  const float* embed     = (const float*)d_in[2];
  const float* start_emb = (const float*)d_in[3];
  const float* proj_h_w  = (const float*)d_in[4];
  const float* proj_h_b  = (const float*)d_in[5];
  const float* proj_c_w  = (const float*)d_in[6];
  const float* proj_c_b  = (const float*)d_in[7];
  const float* w_ih0     = (const float*)d_in[8];
  const float* w_hh0     = (const float*)d_in[9];
  const float* b0        = (const float*)d_in[10];
  const float* w_ih1     = (const float*)d_in[11];
  const float* w_hh1     = (const float*)d_in[12];
  const float* b1        = (const float*)d_in[13];
  const float* fc_w      = (const float*)d_in[14];
  const float* fc_b      = (const float*)d_in[15];

  char* ws = (char*)d_ws;
  size_t o = 0;
  _Float16* Wp0 = (_Float16*)(ws + o); o += (size_t)24 * 64  * 1024;  // 1.5 MB
  _Float16* Wp1 = (_Float16*)(ws + o); o += (size_t)16 * 64  * 1024;  // 1.0 MB
  _Float16* Wfc = (_Float16*)(ws + o); o += (size_t)8  * 500 * 1024;  // 4.0 MB
  _Float16* Wpj = (_Float16*)(ws + o); o += (size_t)16 * 32  * 1024;  // 0.5 MB
  _Float16* ctx_ws = (_Float16*)(ws + o); o += (size_t)64 * H_ * 2;
  float*    c0_ws  = (float*)(ws + o);    o += (size_t)64 * H_ * 4;
  float*    sel_ws = (float*)(ws + o);    o += (size_t)(SEG_ + 1) * NST * 4;
  float*    eos_ws = (float*)(ws + o);    o += (size_t)(SEG_ + 1) * NST * 4;

  // Pack weights into WMMA B-fragment layout (f16)
  pack_kcat<<<(24 * 64 * 32 + 255) / 256, 256, 0, stream>>>(Wp0, w_ih0, w_hh0, 512, 256, 1024);
  pack_kcat<<<(16 * 64 * 32 + 255) / 256, 256, 0, stream>>>(Wp1, w_ih1, w_hh1, 256, 256, 1024);
  pack_kcat<<<(8 * 500 * 32 + 255) / 256, 256, 0, stream>>>(Wfc, fc_w, fc_w, 256, 0, 8000);
  pack_ncat<<<(16 * 32 * 32 + 255) / 256, 256, 0, stream>>>(Wpj, proj_h_w, proj_c_w, 512, 256);

  // h0/c0 projection (WMMA GEMM)
  init_proj_kernel<<<B_, 256, 0, stream>>>(enc, proj_h_b, proj_c_b, Wpj, ctx_ws, c0_ws);

  // 5-step fused LSTM + fc + log-softmax
  decode_kernel<<<NST / 16, 256, 0, stream>>>(ytok, embed, start_emb, b0, b1, fc_b,
                                              Wp0, Wp1, Wfc, ctx_ws, c0_ws,
                                              sel_ws, eos_ws);

  // Final alignment DP -> 4 floats
  dp_kernel<<<1, 96, 0, stream>>>(sel_ws, eos_ws, (float*)d_out);
}
